// DiSAN_25185688223846
// MI455X (gfx1250) — compile-verified, matrix operand order checked by reading, not run on hardware
//
#include <hip/hip_runtime.h>
#include <hip/hip_bf16.h>

typedef __attribute__((ext_vector_type(16))) _Float16 v16h;
typedef __attribute__((ext_vector_type(8)))  float    v8f;

#define BATCH 8
#define SEQL  128
#define DIM   200
#define KD    224   // K pad for D (7*32)
#define D2    400
#define KD2   416   // K pad for 2D (13*32)
#define NSEQ  16    // 2 blocks * 8 batch
#define BL    2048  // NSEQ * SEQL
#define NEGINF (-1.0e13f)

// padded N-tile counts (even, so each wave owns a pair of tiles)
#define NTP_D   14  // ceil(200/16)=13 -> pad to 14
#define NWT_D   7
#define KB_D    7   // KD/32
#define NTP_D2  26  // 400/16 = 25 -> pad to 26
#define NWT_D2  13
#define KB_D2   13  // KD2/32

// WMMA 16-bit operand K mapping (documented A layout; B assumed mirrored with N on lanes):
// lanes 0-15 hold K {0..7,16..23}, lanes 16-31 hold K {8..15,24..31} within a 32-K block.
__device__ __forceinline__ int kmap(int e, int hi) {
  return e + ((e >= 8) ? 8 : 0) + hi * 8;
}

// ---------------- weight f32 -> packed f16 B-fragments ----------------
// Bp layout: fragment (nt, kb) occupies 512 halves at ((nt*KB + kb)*32 + lane)*16.
__global__ void cvtw_pack_k(const float* __restrict__ W, _Float16* __restrict__ Bp,
                            int K, int N, int KB) {
  int idx  = blockIdx.x;        // nt*KB + kb
  int nt   = idx / KB;
  int kb   = idx % KB;
  int lane = threadIdx.x & 31;
  int hi   = lane >> 4;
  int lm   = lane & 15;
  int n    = nt * 16 + lm;
  _Float16* out = Bp + ((size_t)idx * 32 + lane) * 16;
#pragma unroll
  for (int e = 0; e < 16; ++e) {
    int k = kb * 32 + kmap(e, hi);
    float v = (k < K && n < N) ? W[(size_t)k * N + n] : 0.0f;
    out[e] = (_Float16)v;
  }
}

// ---------------- activations f32 -> padded f16 (GEMM A operand) ----------------
__global__ void cvta_k(const float* __restrict__ in, _Float16* __restrict__ out,
                       int N, int Kp) {
  int row = blockIdx.x; // 0..BL-1
  for (int d = threadIdx.x; d < Kp; d += blockDim.x) {
    float v = (d < N) ? in[(size_t)row * N + d] : 0.0f;
    out[(size_t)row * Kp + d] = (_Float16)v;
  }
}

// ---------------- embedding gather -> padded f16 ----------------
__global__ void embed_k(const int* __restrict__ x1, const int* __restrict__ x2,
                        const float* __restrict__ emb, _Float16* __restrict__ x16) {
  int row = blockIdx.x;          // 0..BL-1
  int seq = row >> 7;
  int pos = row & (SEQL - 1);
  int id  = (seq < BATCH) ? x1[seq * SEQL + pos] : x2[(seq - BATCH) * SEQL + pos];
  const float* e = emb + (size_t)id * DIM;
  for (int d = threadIdx.x; d < KD; d += blockDim.x) {
    float v = (d < DIM) ? e[d] : 0.0f;
    x16[(size_t)row * KD + d] = (_Float16)v;
  }
}

// ---------------- WMMA GEMM: C = act(A @ W + bias [+ C]) ----------------
// One wave32 per 16x32 output block (2 accumulators, shared A fragment).
// A: [BL][Kp] f16 row-major. Bp: packed fragments. ACT: 0 none, 1 ELU.
template <int ACT, bool ACCUM>
__global__ void gemm_wmma_k(const _Float16* __restrict__ A, const _Float16* __restrict__ Bp,
                            const float* __restrict__ bias, float* __restrict__ C,
                            int N, int Kp, int KB, int nwt) {
  int m0  = (blockIdx.x / nwt) * 16;
  int wt  = blockIdx.x % nwt;
  int nt0 = wt * 2;
  int nt1 = nt0 + 1;
  int lane = threadIdx.x & 31;
  int hi   = lane >> 4;
  int lm   = lane & 15;

  v8f acc0 = {}, acc1 = {};
  const _Float16* Arow = A + (size_t)(m0 + lm) * Kp;
  const _Float16* B0   = Bp + ((size_t)nt0 * KB) * 512 + (size_t)lane * 16;
  const _Float16* B1   = Bp + ((size_t)nt1 * KB) * 512 + (size_t)lane * 16;

  for (int kb = 0; kb < KB; ++kb) {
    v16h a;
#pragma unroll
    for (int e = 0; e < 16; ++e) a[e] = Arow[kb * 32 + kmap(e, hi)];
    v16h b0 = *(const v16h*)(B0 + (size_t)kb * 512);
    v16h b1 = *(const v16h*)(B1 + (size_t)kb * 512);
    if (kb + 1 < KB) {  // lowers to global_prefetch_b8
      __builtin_prefetch((const void*)(B0 + (size_t)(kb + 1) * 512), 0, 0);
      __builtin_prefetch((const void*)(B1 + (size_t)(kb + 1) * 512), 0, 0);
    }
    acc0 = __builtin_amdgcn_wmma_f32_16x16x32_f16(false, a, false, b0, (short)0, acc0,
                                                  false, false);
    acc1 = __builtin_amdgcn_wmma_f32_16x16x32_f16(false, a, false, b1, (short)0, acc1,
                                                  false, false);
  }

  int rbase = m0 + hi * 8;   // C layout: VGPR r -> row r (lanes 0-15) / r+8 (lanes 16-31)
#pragma unroll
  for (int t = 0; t < 2; ++t) {
    int col = (t ? nt1 : nt0) * 16 + lm;
    if (col >= N) continue;
    v8f acc = t ? acc1 : acc0;
#pragma unroll
    for (int r = 0; r < 8; ++r) {
      size_t idx = (size_t)(rbase + r) * N + col;
      float v = acc[r];
      if (ACCUM) v += C[idx];
      if (bias)  v += bias[col];
      if (ACT == 1) v = (v > 0.0f) ? v : (__expf(v) - 1.0f);
      C[idx] = v;
    }
  }
}

// ---------------- fused directional multi-dim attention ----------------
// One block per (seq, l) query row; thread d owns feature dim d; streaming softmax
// over keys m for both directions simultaneously. (B,L,L,D) tensor never materialized.
__global__ void attn_k(const float* __restrict__ h, const float* __restrict__ h1,
                       const float* __restrict__ h2, const float* __restrict__ bvec,
                       const float* __restrict__ cptr,
                       const int* __restrict__ x1, const int* __restrict__ x2,
                       float* __restrict__ s_fw, float* __restrict__ s_bw) {
  int row = blockIdx.x;
  int seq = row >> 7;
  int l   = row & (SEQL - 1);
  __shared__ int   padm[SEQL];
  __shared__ float cval;
  const int* ids = (seq < BATCH) ? (x1 + seq * SEQL) : (x2 + (seq - BATCH) * SEQL);
  for (int m = threadIdx.x; m < SEQL; m += blockDim.x) padm[m] = (ids[m] == 1);
  if (threadIdx.x == 0) cval = cptr[0];
  __syncthreads();

  int d = threadIdx.x;
  if (d >= DIM) return;
  float c     = cval;
  float rcpc  = 1.0f / c;
  int   pad_l = padm[l];
  float h1v   = h1[(size_t)row * DIM + d] + bvec[d];
  const float* h2s = h2 + (size_t)(seq * SEQL) * DIM + d;
  const float* hs  = h  + (size_t)(seq * SEQL) * DIM + d;

  float mxf = -3.0e38f, smf = 0.0f, acf = 0.0f;
  float mxb = -3.0e38f, smb = 0.0f, acb = 0.0f;
  for (int m = 0; m < SEQL; ++m) {
    float att = c * tanhf((h1v + h2s[(size_t)m * DIM]) * rcpc);
    if (pad_l || padm[m]) att = NEGINF;
    float hv  = hs[(size_t)m * DIM];
    float afw = (m <= l) ? NEGINF : att;   // fw masks j<=i (tril)
    float abw = (m >= l) ? NEGINF : att;   // bw masks j>=i (triu)
    if (afw > mxf) { float sc = __expf(mxf - afw); smf *= sc; acf *= sc; mxf = afw; }
    float ef = __expf(afw - mxf); smf += ef; acf += ef * hv;
    if (abw > mxb) { float sc = __expf(mxb - abw); smb *= sc; acb *= sc; mxb = abw; }
    float eb = __expf(abw - mxb); smb += eb; acb += eb * hv;
  }
  s_fw[(size_t)row * DIM + d] = acf / smf;
  s_bw[(size_t)row * DIM + d] = acb / smb;
}

// ---------------- sigmoid gate fuse: u = [f*h+(1-f)*s_fw , f*h+(1-f)*s_bw] ----------------
__global__ void fuse_k(const float* __restrict__ h, const float* __restrict__ sfw,
                       const float* __restrict__ sbw, const float* __restrict__ fbf,
                       const float* __restrict__ fbb, float* __restrict__ u) {
  int row = blockIdx.x;
  for (int d = threadIdx.x; d < DIM; d += blockDim.x) {
    size_t i = (size_t)row * DIM + d;
    float hv = h[i];
    float ff = 1.0f / (1.0f + __expf(-fbf[i]));
    float fb = 1.0f / (1.0f + __expf(-fbb[i]));
    u[(size_t)row * D2 + d]       = ff * hv + (1.0f - ff) * sfw[i];
    u[(size_t)row * D2 + DIM + d] = fb * hv + (1.0f - fb) * sbw[i];
  }
}

// ---------------- per-sequence weighted reduce: bo[seq,e] = sum_l u*att_s ----------------
__global__ void reduce_k(const float* __restrict__ u, const float* __restrict__ atts,
                         float* __restrict__ bo) {
  int seq = blockIdx.x;
  for (int e = threadIdx.x; e < D2; e += blockDim.x) {
    float s = 0.0f;
    for (int l = 0; l < SEQL; ++l) {
      size_t i = (size_t)(seq * SEQL + l) * D2 + e;
      s += u[i] * atts[i];
    }
    bo[seq * D2 + e] = s;
  }
}

// ---------------- head: y = relu(feat @ F1 + b) @ F2 + b ----------------
__global__ void head_k(const float* __restrict__ bo, const float* __restrict__ F1,
                       const float* __restrict__ F1b, const float* __restrict__ F2,
                       const float* __restrict__ F2b, float* __restrict__ y) {
  int b = blockIdx.x;                    // 0..7
  const float* cv = bo + b * D2;
  const float* rv = bo + (BATCH + b) * D2;
  __shared__ float red[256];
  float part = 0.0f;
  for (int j = threadIdx.x; j < DIM; j += blockDim.x) {
    float hsum = F1b[j];
    for (int k = 0; k < D2; ++k) {
      float cvk = cv[k], rvk = rv[k];
      hsum += cvk         * F1[(size_t)(k)        * DIM + j];
      hsum += rvk         * F1[(size_t)(D2 + k)   * DIM + j];
      hsum += (cvk - rvk) * F1[(size_t)(2*D2 + k) * DIM + j];
      hsum += (cvk * rvk) * F1[(size_t)(3*D2 + k) * DIM + j];
    }
    hsum = fmaxf(hsum, 0.0f);
    part += hsum * F2[j];
  }
  red[threadIdx.x] = part;
  __syncthreads();
  for (int s = blockDim.x >> 1; s > 0; s >>= 1) {
    if ((int)threadIdx.x < s) red[threadIdx.x] += red[threadIdx.x + s];
    __syncthreads();
  }
  if (threadIdx.x == 0) y[b] = red[0] + F2b[0];
}

// ---------------- host launch ----------------
extern "C" void kernel_launch(void* const* d_in, const int* in_sizes, int n_in,
                              void* d_out, int out_size, void* d_ws, size_t ws_size,
                              hipStream_t stream) {
  const int*   x1    = (const int*)d_in[0];
  const int*   x2    = (const int*)d_in[1];
  const float* emb   = (const float*)d_in[2];
  const float* Wh_w  = (const float*)d_in[3];
  const float* Wh_b  = (const float*)d_in[4];
  const float* W1_w  = (const float*)d_in[5];
  const float* W2_w  = (const float*)d_in[6];
  const float* bvec  = (const float*)d_in[7];
  const float* cptr  = (const float*)d_in[8];
  const float* Wf1_w = (const float*)d_in[9];
  const float* Wf2_w = (const float*)d_in[10];
  const float* Wf2_b = (const float*)d_in[11];
  const float* Ws1_w = (const float*)d_in[12];
  const float* Ws1_b = (const float*)d_in[13];
  const float* Ws_w  = (const float*)d_in[14];
  const float* Ws_b  = (const float*)d_in[15];
  const float* F1_w  = (const float*)d_in[16];
  const float* F1_b  = (const float*)d_in[17];
  const float* F2_w  = (const float*)d_in[18];
  const float* F2_b  = (const float*)d_in[19];
  float* y = (float*)d_out;

  char*  base = (char*)d_ws;
  size_t off  = 0;
  auto alloc = [&](size_t bytes) -> void* {
    void* p = base + off;
    off = (off + bytes + 255) & ~(size_t)255;
    return p;
  };

  const size_t packD  = (size_t)NTP_D  * KB_D  * 512 * 2;  // packed D-weight bytes
  const size_t packD2 = (size_t)NTP_D2 * KB_D2 * 512 * 2;  // packed 2D-weight bytes

  _Float16* x16   = (_Float16*)alloc((size_t)BL * KD * 2);
  float*    h     = (float*)   alloc((size_t)BL * DIM * 4);
  _Float16* h16   = (_Float16*)alloc((size_t)BL * KD * 2);
  float*    h1    = (float*)   alloc((size_t)BL * DIM * 4);
  float*    h2    = (float*)   alloc((size_t)BL * DIM * 4);
  float*    s_fw  = (float*)   alloc((size_t)BL * DIM * 4);
  float*    s_bw  = (float*)   alloc((size_t)BL * DIM * 4);
  _Float16* sfw16 = (_Float16*)alloc((size_t)BL * KD * 2);
  _Float16* sbw16 = (_Float16*)alloc((size_t)BL * KD * 2);
  float*    fb_fw = (float*)   alloc((size_t)BL * DIM * 4);
  float*    fb_bw = (float*)   alloc((size_t)BL * DIM * 4);
  float*    u     = (float*)   alloc((size_t)BL * D2 * 4);
  _Float16* u16   = (_Float16*)alloc((size_t)BL * KD2 * 2);
  float*    t1    = (float*)   alloc((size_t)BL * D2 * 4);
  _Float16* t116  = (_Float16*)alloc((size_t)BL * KD2 * 2);
  float*    atts  = (float*)   alloc((size_t)BL * D2 * 4);
  float*    bo    = (float*)   alloc((size_t)NSEQ * D2 * 4);
  _Float16* Wh16  = (_Float16*)alloc(packD);
  _Float16* W116  = (_Float16*)alloc(packD);
  _Float16* W216  = (_Float16*)alloc(packD);
  _Float16* Wf116 = (_Float16*)alloc(packD);
  _Float16* Wf216 = (_Float16*)alloc(packD);
  _Float16* Ws116 = (_Float16*)alloc(packD2);
  _Float16* Ws16  = (_Float16*)alloc(packD2);

  // 1. weights -> packed f16 WMMA B-fragments
  cvtw_pack_k<<<NTP_D  * KB_D,  32, 0, stream>>>(Wh_w,  Wh16,  DIM, DIM, KB_D);
  cvtw_pack_k<<<NTP_D  * KB_D,  32, 0, stream>>>(W1_w,  W116,  DIM, DIM, KB_D);
  cvtw_pack_k<<<NTP_D  * KB_D,  32, 0, stream>>>(W2_w,  W216,  DIM, DIM, KB_D);
  cvtw_pack_k<<<NTP_D  * KB_D,  32, 0, stream>>>(Wf1_w, Wf116, DIM, DIM, KB_D);
  cvtw_pack_k<<<NTP_D  * KB_D,  32, 0, stream>>>(Wf2_w, Wf216, DIM, DIM, KB_D);
  cvtw_pack_k<<<NTP_D2 * KB_D2, 32, 0, stream>>>(Ws1_w, Ws116, D2,  D2,  KB_D2);
  cvtw_pack_k<<<NTP_D2 * KB_D2, 32, 0, stream>>>(Ws_w,  Ws16,  D2,  D2,  KB_D2);

  // 2. embedding gather (both blocks, 2048 rows)
  embed_k<<<BL, 256, 0, stream>>>(x1, x2, emb, x16);

  const int gridD  = (BL / 16) * NWT_D;   // 128 * 7
  const int gridD2 = (BL / 16) * NWT_D2;  // 128 * 13

  // 3. h = elu(x @ Wh + bh)
  gemm_wmma_k<1, false><<<gridD, 32, 0, stream>>>(x16, Wh16, Wh_b, h, DIM, KD, KB_D, NWT_D);
  cvta_k<<<BL, 256, 0, stream>>>(h, h16, DIM, KD);

  // 4. h1 = h @ W1 ; h2 = h @ W2
  gemm_wmma_k<0, false><<<gridD, 32, 0, stream>>>(h16, W116, nullptr, h1, DIM, KD, KB_D, NWT_D);
  gemm_wmma_k<0, false><<<gridD, 32, 0, stream>>>(h16, W216, nullptr, h2, DIM, KD, KB_D, NWT_D);

  // 5. streaming directional multi-dim attention (fw + bw in one pass)
  attn_k<<<BL, 256, 0, stream>>>(h, h1, h2, bvec, cptr, x1, x2, s_fw, s_bw);
  cvta_k<<<BL, 256, 0, stream>>>(s_fw, sfw16, DIM, KD);
  cvta_k<<<BL, 256, 0, stream>>>(s_bw, sbw16, DIM, KD);

  // 6. gate logits: fb = h @ Wf2 + bf2 ; fb += s @ Wf1
  gemm_wmma_k<0, false><<<gridD, 32, 0, stream>>>(h16, Wf216, Wf2_b, fb_fw, DIM, KD, KB_D, NWT_D);
  gemm_wmma_k<0, true ><<<gridD, 32, 0, stream>>>(sfw16, Wf116, nullptr, fb_fw, DIM, KD, KB_D, NWT_D);
  gemm_wmma_k<0, false><<<gridD, 32, 0, stream>>>(h16, Wf216, Wf2_b, fb_bw, DIM, KD, KB_D, NWT_D);
  gemm_wmma_k<0, true ><<<gridD, 32, 0, stream>>>(sbw16, Wf116, nullptr, fb_bw, DIM, KD, KB_D, NWT_D);

  // 7. u = [gate(fw), gate(bw)]
  fuse_k<<<BL, 256, 0, stream>>>(h, s_fw, s_bw, fb_fw, fb_bw, u);
  cvta_k<<<BL, 256, 0, stream>>>(u, u16, D2, KD2);

  // 8. t1 = elu(u @ Ws1 + b) ; att_s = t1 @ Ws + b
  gemm_wmma_k<1, false><<<gridD2, 32, 0, stream>>>(u16, Ws116, Ws1_b, t1, D2, KD2, KB_D2, NWT_D2);
  cvta_k<<<BL, 256, 0, stream>>>(t1, t116, D2, KD2);
  gemm_wmma_k<0, false><<<gridD2, 32, 0, stream>>>(t116, Ws16, Ws_b, atts, D2, KD2, KB_D2, NWT_D2);

  // 9. per-sequence reduce, then head
  reduce_k<<<NSEQ, 256, 0, stream>>>(u, atts, bo);
  head_k<<<BATCH, 256, 0, stream>>>(bo, F1_w, F1_b, F2_w, F2_b, y);

  (void)in_sizes; (void)n_in; (void)out_size; (void)ws_size;
}
